// GraphCNN_44968307589411
// MI455X (gfx1250) — matrix operand order repaired and checked by reading, not compiled, not run on hardware
//
#include <hip/hip_runtime.h>

#define N_NODES 20000
#define N_EDGES 320000
#define DIM     512
#define LAYERS  4

typedef float          v8f   __attribute__((ext_vector_type(8)));
typedef __bf16         v16bf __attribute__((ext_vector_type(16)));
typedef unsigned short u16x8 __attribute__((ext_vector_type(8)));
typedef unsigned short u16x4 __attribute__((ext_vector_type(4)));

union FragBF { v16bf v; u16x8 h[2]; };

__device__ __forceinline__ unsigned short f32_to_bf16(float f) {
  unsigned int u = __float_as_uint(f);
  u += 0x7FFFu + ((u >> 16) & 1u);          // round-to-nearest-even on raw bits
  return (unsigned short)(u >> 16);
}
__device__ __forceinline__ float bf16_to_f32(unsigned short h) {
  return __uint_as_float(((unsigned int)h) << 16);
}

// ---------------------------------------------------------------- utilities
__global__ void zero_f32_kernel(float* p, int n4) {
  int i = blockIdx.x * blockDim.x + threadIdx.x;
  if (i < n4) ((float4*)p)[i] = make_float4(0.f, 0.f, 0.f, 0.f);
}

__global__ void f32_to_bf16_kernel(const float* __restrict__ s,
                                   unsigned short* __restrict__ d, int n4) {
  int i = blockIdx.x * blockDim.x + threadIdx.x;
  if (i >= n4) return;
  float4 v = ((const float4*)s)[i];
  u16x4 o = { f32_to_bf16(v.x), f32_to_bf16(v.y), f32_to_bf16(v.z), f32_to_bf16(v.w) };
  ((u16x4*)d)[i] = o;
}

// ---------------------------------------------------------------- degree
__global__ void deg_count_kernel(const int* __restrict__ dst, float* __restrict__ deg, int e) {
  int i = blockIdx.x * blockDim.x + threadIdx.x;
  if (i < e) atomicAdd(&deg[dst[i]], 1.0f);
}
__global__ void inv_deg_kernel(float* deg, int n) {
  int i = blockIdx.x * blockDim.x + threadIdx.x;
  if (i < n) deg[i] = 1.0f / fmaxf(deg[i], 1.0f);
}

// ---------------------------------------------------------------- embedding
__global__ void embed_kernel(const int* __restrict__ tok, const int* __restrict__ pos,
                             const float* __restrict__ ve, const float* __restrict__ pe,
                             float* __restrict__ x) {
  int t = blockIdx.x * blockDim.x + threadIdx.x;     // N*D/4 threads
  int node = t >> 7;                                  // D/4 = 128 per node
  int c    = (t & 127) << 2;
  float4 a = *(const float4*)(ve + tok[node] * DIM + c);
  float4 b = *(const float4*)(pe + pos[node] * DIM + c);
  float4 o = make_float4(a.x + b.x, a.y + b.y, a.z + b.z, a.w + b.w);
  *(float4*)(x + node * DIM + c) = o;
}

// ---------------------------------------------------------------- batchnorm -> bf16
__global__ void bn_kernel(const float* __restrict__ x,
                          const float* __restrict__ gamma, const float* __restrict__ beta,
                          const float* __restrict__ mean,  const float* __restrict__ var,
                          unsigned short* __restrict__ xn) {
  int t = blockIdx.x * blockDim.x + threadIdx.x;     // N*D/4 threads
  int node = t >> 7;
  int c    = (t & 127) << 2;
  float4 v = *(const float4*)(x + node * DIM + c);
  u16x4 o;
  float f[4] = { v.x, v.y, v.z, v.w };
#pragma unroll
  for (int j = 0; j < 4; ++j) {
    int col = c + j;
    float inv = gamma[col] * rsqrtf(var[col] + 1e-5f);
    o[j] = f32_to_bf16((f[j] - mean[col]) * inv + beta[col]);
  }
  *(u16x4*)(xn + node * DIM + c) = o;
}

// ---------------------------------------------------------------- edge scatter (sum)
__global__ void scatter_kernel(const unsigned short* __restrict__ xn,
                               const int* __restrict__ src, const int* __restrict__ dst,
                               float* __restrict__ agg) {
  int tid = blockIdx.x * blockDim.x + threadIdx.x;   // E * 64 threads
  int e = tid >> 6;
  int c = (tid & 63) << 3;                           // 8 elems per thread
  int s = src[e], d = dst[e];
  u16x8 v = *(const u16x8*)(xn + (size_t)s * DIM + c);
  float* out = agg + (size_t)d * DIM + c;
#pragma unroll
  for (int i = 0; i < 8; ++i) atomicAdd(out + i, bf16_to_f32(v[i]));
}

// ---------------------------------------------------------------- fused dual-GEMM + bias + ReLU
//   out[16 x 512 strip] = relu( (agg/deg) @ Wrel^T + xn @ Wroot^T + b )
__global__ __launch_bounds__(256) void gemm_fused_kernel(
    const float* __restrict__ agg, const unsigned short* __restrict__ xn,
    const unsigned short* __restrict__ Wrel, const unsigned short* __restrict__ Wroot,
    const float* __restrict__ bias, const float* __restrict__ invdeg,
    float* __restrict__ out) {
  __shared__ __align__(16) unsigned short ldsAgg[16 * DIM];
  __shared__ __align__(16) unsigned short ldsXn[16 * DIM];

  const int rowBase = blockIdx.x * 16;
  const int tid = threadIdx.x;

  // Stage the 16 x 512 A-strips into LDS (xn copied, agg scaled by 1/deg + cvt bf16)
#pragma unroll
  for (int it = 0; it < 4; ++it) {
    int chunk = tid + it * 256;          // 1024 chunks of 8 elements
    int off = chunk << 3;
    int row = off >> 9;                  // /512
    int gidx = (rowBase + row) * DIM + (off & (DIM - 1));
    *(u16x8*)(ldsXn + off) = *(const u16x8*)(xn + gidx);
    float sc = invdeg[rowBase + row];
    float4 f0 = *(const float4*)(agg + gidx);
    float4 f1 = *(const float4*)(agg + gidx + 4);
    u16x8 o;
    o[0] = f32_to_bf16(f0.x * sc); o[1] = f32_to_bf16(f0.y * sc);
    o[2] = f32_to_bf16(f0.z * sc); o[3] = f32_to_bf16(f0.w * sc);
    o[4] = f32_to_bf16(f1.x * sc); o[5] = f32_to_bf16(f1.y * sc);
    o[6] = f32_to_bf16(f1.z * sc); o[7] = f32_to_bf16(f1.w * sc);
    *(u16x8*)(ldsAgg + off) = o;
  }
  __syncthreads();

  const int lane = tid & 31;
  const int wave = tid >> 5;
  const int half = lane >> 4;            // ISA lane-half
  const int mrow = lane & 15;            // A row within tile (also D/C column)
  const int colBase = wave * 64;         // each wave: 4 x 16 output columns

  v8f acc[4];
#pragma unroll
  for (int t = 0; t < 4; ++t)
#pragma unroll
    for (int r = 0; r < 8; ++r) acc[t][r] = 0.0f;

  for (int ks = 0; ks < DIM / 32; ++ks) {
    const int k0 = ks * 32;
    // A frags from LDS: lane half 0 -> K 0..15; half 1 -> K 16..31
    const int abase = mrow * DIM + k0 + 16 * half;
    FragBF aAgg, aXn;
    aAgg.h[0] = *(const u16x8*)(ldsAgg + abase);
    aAgg.h[1] = *(const u16x8*)(ldsAgg + abase + 8);
    aXn.h[0]  = *(const u16x8*)(ldsXn + abase);
    aXn.h[1]  = *(const u16x8*)(ldsXn + abase + 8);

#pragma unroll
    for (int t = 0; t < 4; ++t) {
      // B frag: B[d,n] = W[n,d]; lanes 0-15 hold K=0..15, lanes 16-31 hold K=16..31
      const int wrow = colBase + t * 16 + mrow;
      const unsigned short* br = Wrel  + (size_t)wrow * DIM + k0 + 16 * half;
      const unsigned short* bo = Wroot + (size_t)wrow * DIM + k0 + 16 * half;
      FragBF bRel, bRoot;
      bRel.h[0]  = *(const u16x8*)br;       bRel.h[1]  = *(const u16x8*)(br + 8);
      bRoot.h[0] = *(const u16x8*)bo;       bRoot.h[1] = *(const u16x8*)(bo + 8);
      __builtin_prefetch(br + 32, 0, 3);    // next K-block of this W row
      __builtin_prefetch(bo + 32, 0, 3);
      acc[t] = __builtin_amdgcn_wmma_f32_16x16x32_bf16(
          false, aAgg.v, false, bRel.v,  (short)0, acc[t], false, false);
      acc[t] = __builtin_amdgcn_wmma_f32_16x16x32_bf16(
          false, aXn.v,  false, bRoot.v, (short)0, acc[t], false, false);
    }
  }

  // Epilogue: C/D layout -> M = r + 8*half, N = lane&15
#pragma unroll
  for (int t = 0; t < 4; ++t) {
    int n = colBase + t * 16 + mrow;
    float b = bias[n];
#pragma unroll
    for (int r = 0; r < 8; ++r) {
      int m = r + 8 * half;
      float v = acc[t][r] + b;
      out[(size_t)(rowBase + m) * DIM + n] = v > 0.0f ? v : 0.0f;
    }
  }
}

// ---------------------------------------------------------------- launch
extern "C" void kernel_launch(void* const* d_in, const int* in_sizes, int n_in,
                              void* d_out, int out_size, void* d_ws, size_t ws_size,
                              hipStream_t stream) {
  (void)in_sizes; (void)n_in; (void)out_size; (void)ws_size;
  const int*   tokens   = (const int*)d_in[0];
  const int*   pos      = (const int*)d_in[1];
  const int*   edge     = (const int*)d_in[2];   // [2,E]: src then dst
  const float* vocabEmb = (const float*)d_in[3];
  const float* posEmb   = (const float*)d_in[4];
  const float* bnGamma  = (const float*)d_in[5];
  const float* bnBeta   = (const float*)d_in[6];
  const float* bnMean   = (const float*)d_in[7];
  const float* bnVar    = (const float*)d_in[8];
  const float* Wrel     = (const float*)d_in[9];
  const float* bRel     = (const float*)d_in[10];
  const float* Wroot    = (const float*)d_in[11];
  float* out = (float*)d_out;

  char* ws = (char*)d_ws;
  size_t off = 0;
  auto alloc = [&](size_t bytes) -> void* {
    void* p = ws + off;
    off += (bytes + 255) & ~(size_t)255;
    return p;
  };
  float*          x      = (float*)alloc((size_t)N_NODES * DIM * 4);
  float*          agg    = (float*)alloc((size_t)N_NODES * DIM * 4);
  unsigned short* xn     = (unsigned short*)alloc((size_t)N_NODES * DIM * 2);
  unsigned short* wrelB  = (unsigned short*)alloc((size_t)LAYERS * DIM * DIM * 2);
  unsigned short* wrootB = (unsigned short*)alloc((size_t)LAYERS * DIM * DIM * 2);
  float*          deg    = (float*)alloc((size_t)N_NODES * 4);

  const int THREADS = 256;
  const int nd4   = N_NODES * DIM / 4;          // 2,560,000
  const int ndGrid = nd4 / THREADS;             // 10,000

  // degrees -> 1/max(deg,1)
  zero_f32_kernel<<<(N_NODES / 4 + THREADS - 1) / THREADS, THREADS, 0, stream>>>(deg, N_NODES / 4);
  deg_count_kernel<<<(N_EDGES + THREADS - 1) / THREADS, THREADS, 0, stream>>>(edge + N_EDGES, deg, N_EDGES);
  inv_deg_kernel<<<(N_NODES + THREADS - 1) / THREADS, THREADS, 0, stream>>>(deg, N_NODES);

  // weights -> bf16 (all layers at once)
  const int w4 = LAYERS * DIM * DIM / 4;        // 262,144
  f32_to_bf16_kernel<<<w4 / THREADS, THREADS, 0, stream>>>(Wrel, wrelB, w4);
  f32_to_bf16_kernel<<<w4 / THREADS, THREADS, 0, stream>>>(Wroot, wrootB, w4);

  // x = vocab_emb[tokens] + pos_emb[pos]
  embed_kernel<<<ndGrid, THREADS, 0, stream>>>(tokens, pos, vocabEmb, posEmb, x);

  for (int l = 0; l < LAYERS; ++l) {
    bn_kernel<<<ndGrid, THREADS, 0, stream>>>(x, bnGamma + l * DIM, bnBeta + l * DIM,
                                              bnMean + l * DIM, bnVar + l * DIM, xn);
    zero_f32_kernel<<<ndGrid, THREADS, 0, stream>>>(agg, nd4);
    scatter_kernel<<<N_EDGES * 64 / THREADS, THREADS, 0, stream>>>(xn, edge, edge + N_EDGES, agg);
    gemm_fused_kernel<<<N_NODES / 16, THREADS, 0, stream>>>(
        agg, xn, wrelB + (size_t)l * DIM * DIM, wrootB + (size_t)l * DIM * DIM,
        bRel + l * DIM, deg, (l == LAYERS - 1) ? out : x);
  }
}
